// SparseAttention_76914274336769
// MI455X (gfx1250) — matrix operand order repaired and checked by reading, not compiled, not run on hardware
//
#include <hip/hip_runtime.h>

typedef _Float16 half_t;
typedef __attribute__((ext_vector_type(16))) _Float16 v16h;
typedef __attribute__((ext_vector_type(8)))  _Float16 v8h;
typedef __attribute__((ext_vector_type(8)))  float    v8f;
typedef __attribute__((ext_vector_type(4)))  float    v4f;
typedef __attribute__((ext_vector_type(4)))  int      v4i;

#define SEQ   2048
#define HDIM  1024
#define NHEAD 16
#define DHEAD 64
#define KTOP  64

__device__ __forceinline__ v8f wmma_f16(v16h a, v16h b, v8f c) {
  // (neg_a, A, neg_b, B, c_mod, C, reuse_a, reuse_b)
  return __builtin_amdgcn_wmma_f32_16x16x32_f16(false, a, false, b, (short)0, c,
                                                false, false);
}

// A operand (16x32 f16): lane holds row m=lane%16; elements 0..7 = K c0..c0+7,
// elements 8..15 = K c0+16..c0+23; caller passes c0 = base + 8*(lane>=16).
__device__ __forceinline__ v16h make_a_f16(const half_t* row, int c0) {
  v8h lo = *(const v8h*)(row + c0);
  v8h hi = *(const v8h*)(row + c0 + 16);
  v16h a;
#pragma unroll
  for (int i = 0; i < 8; ++i) { a[i] = lo[i]; a[8 + i] = hi[i]; }
  return a;
}

__device__ __forceinline__ v16h make_a_f32(const float* row, int c0) {
  v4f x0 = *(const v4f*)(row + c0);
  v4f x1 = *(const v4f*)(row + c0 + 4);
  v4f x2 = *(const v4f*)(row + c0 + 16);
  v4f x3 = *(const v4f*)(row + c0 + 20);
  v16h a;
#pragma unroll
  for (int i = 0; i < 4; ++i) {
    a[i]      = (half_t)x0[i];
    a[4 + i]  = (half_t)x1[i];
    a[8 + i]  = (half_t)x2[i];
    a[12 + i] = (half_t)x3[i];
  }
  return a;
}

// B operand (32x16 f16): lane holds column n=lane%16; elements 0..7 = K
// cb..cb+7, elements 8..15 = K cb+16..cb+23; caller passes cb = base + 8*(lane>=16).
__device__ __forceinline__ v16h make_b_f16(const half_t* row, int cb) {
  v8h lo = *(const v8h*)(row + cb);
  v8h hi = *(const v8h*)(row + cb + 16);
  v16h b;
#pragma unroll
  for (int i = 0; i < 8; ++i) { b[i] = lo[i]; b[8 + i] = hi[i]; }
  return b;
}

// --- CDNA5 async copy to LDS (guarded; falls back to a plain LDS store) -----
__device__ __forceinline__ void async_copy16(half_t* dst_lds,
                                             const half_t* src_g) {
#if defined(__has_builtin) && __has_builtin(__builtin_amdgcn_global_load_async_to_lds_b128)
  __builtin_amdgcn_global_load_async_to_lds_b128(
      (__attribute__((address_space(1))) v4i*)src_g,
      (__attribute__((address_space(3))) v4i*)dst_lds, 0, 0);
#else
  *(v8h*)dst_lds = *(const v8h*)src_g;
#endif
}

__device__ __forceinline__ void wait_async() {
#if defined(__has_builtin) && __has_builtin(__builtin_amdgcn_s_wait_asynccnt)
  __builtin_amdgcn_s_wait_asynccnt(0);
#else
  asm volatile("s_wait_asynccnt 0" ::: "memory");
#endif
}

// ---------------------------------------------------------------------------
// Kernel 0: f32 -> f16 conversion (8 elements/thread, vectorized).
// ---------------------------------------------------------------------------
__global__ void cvt_f16_kernel(const float* __restrict__ src,
                               half_t* __restrict__ dst, int n8) {
  int i = blockIdx.x * blockDim.x + threadIdx.x;
  if (i >= n8) return;
  const float* s = src + (size_t)i * 8;
  v4f x0 = *(const v4f*)(s);
  v4f x1 = *(const v4f*)(s + 4);
  v8h o;
#pragma unroll
  for (int j = 0; j < 4; ++j) { o[j] = (half_t)x0[j]; o[4 + j] = (half_t)x1[j]; }
  *(v8h*)(dst + (size_t)i * 8) = o;
}

// ---------------------------------------------------------------------------
// Kernel 1: QKV projections from f16 operands. C[m,n] = sum_h X[m,h]*W[n,h].
// grid = (H/64, M/64, 3); block = 128 (4 waves, each a 16x64 strip).
// z=0 -> Q (f16 row-major), z=1 -> K (row-major), z=2 -> V per-head transposed
// Vt[b][head][d][s].
// ---------------------------------------------------------------------------
__global__ void proj_qkv_kernel(const half_t* __restrict__ xh,
                                const half_t* __restrict__ Wqh,
                                const half_t* __restrict__ Wkh,
                                const half_t* __restrict__ Wvh,
                                half_t* __restrict__ Qh,
                                half_t* __restrict__ Kh,
                                half_t* __restrict__ Vt) {
  const int lane    = threadIdx.x & 31;
  const int wave    = threadIdx.x >> 5;
  const int half_id = lane >> 4;
  const int ln      = lane & 15;
  const int z       = blockIdx.z;
  const half_t* W = (z == 0) ? Wqh : (z == 1) ? Wkh : Wvh;

  const int mbase = blockIdx.y * 64 + wave * 16;
  const int nbase = blockIdx.x * 64;

  const half_t* xrow = xh + (size_t)(mbase + ln) * HDIM;
  v8f acc[4] = {};

  for (int hb = 0; hb < HDIM; hb += 32) {
    __builtin_prefetch(xrow + hb + 64, 0, 0);
    v16h a = make_a_f16(xrow, hb + half_id * 8);
#pragma unroll
    for (int t = 0; t < 4; ++t) {
      const half_t* wrow = W + (size_t)(nbase + 16 * t + ln) * HDIM;
      v16h b = make_b_f16(wrow, hb + half_id * 8);
      acc[t] = wmma_f16(a, b, acc[t]);
    }
  }

  if (z < 2) {
    half_t* O = (z == 0) ? Qh : Kh;
#pragma unroll
    for (int t = 0; t < 4; ++t) {
      int n = nbase + 16 * t + ln;
#pragma unroll
      for (int r = 0; r < 8; ++r) {
        int mm = mbase + r + 8 * half_id;        // D layout: M = r + 8*(lane>=16)
        O[(size_t)mm * HDIM + n] = (half_t)acc[t][r];
      }
    }
  } else {
    const int head = blockIdx.x;                 // nbase/64
#pragma unroll
    for (int t = 0; t < 4; ++t) {
      int d = 16 * t + ln;
#pragma unroll
      for (int r = 0; r < 8; ++r) {
        int mm = mbase + r + 8 * half_id;
        int bb = mm >> 11;                       // row / SEQ
        int ss = mm & (SEQ - 1);
        Vt[((size_t)((bb * NHEAD + head) * DHEAD + d)) * SEQ + ss] =
            (half_t)acc[t][r];
      }
    }
  }
}

// ---------------------------------------------------------------------------
// Kernel 2: fused sparse attention for one (b,head) and a 16-query tile.
// grid = (SEQ/16, B*NHEAD); block = 128 (4 waves).
// Dynamic LDS: float sc[16][SEQ] (scores -> probs) + half vt[64][32].
// ---------------------------------------------------------------------------
__global__ void sparse_attn_kernel(const half_t* __restrict__ Qh,
                                   const half_t* __restrict__ Kh,
                                   const half_t* __restrict__ Vt,
                                   half_t* __restrict__ AO) {
  extern __shared__ char smem[];
  float*  sc = (float*)smem;                                  // 16*SEQ f32
  half_t* vt = (half_t*)(smem + 16 * SEQ * sizeof(float));    // 64*32 f16

  const int lane    = threadIdx.x & 31;
  const int wave    = threadIdx.x >> 5;
  const int half_id = lane >> 4;
  const int ln      = lane & 15;
  const int bh      = blockIdx.y;
  const int b       = bh >> 4;
  const int head    = bh & 15;
  const int qbase   = blockIdx.x * 16;

  // Preload the q-tile A operands (two 32-wide K chunks over d=0..63).
  const half_t* qrow =
      Qh + (size_t)(b * SEQ + qbase + ln) * HDIM + head * DHEAD;
  v16h a0 = make_a_f16(qrow, 0  + half_id * 8);
  v16h a1 = make_a_f16(qrow, 32 + half_id * 8);

  // Phase 1: scores = q @ k^T / 8 into LDS. Waves split the 128 k-tiles.
  for (int kt = wave; kt < SEQ / 16; kt += 4) {
    const half_t* krow =
        Kh + (size_t)(b * SEQ + kt * 16 + ln) * HDIM + head * DHEAD;
    v16h b0 = make_b_f16(krow, half_id * 8);
    v16h b1 = make_b_f16(krow, 32 + half_id * 8);
    v8f c = {};
    c = wmma_f16(a0, b0, c);
    c = wmma_f16(a1, b1, c);
#pragma unroll
    for (int r = 0; r < 8; ++r)
      sc[(size_t)(r + 8 * half_id) * SEQ + kt * 16 + ln] = c[r] * 0.125f;
  }
  __syncthreads();

  // Phase 1.5: per-row top-64 threshold (bitwise radix-select on an order-
  // preserving uint key), then rewrite the row in place with softmax probs.
  for (int m = wave; m < 16; m += 4) {
    float* row = sc + (size_t)m * SEQ;
    float rmax = -3.4e38f;
    for (int j = lane; j < SEQ; j += 32) rmax = fmaxf(rmax, row[j]);
#pragma unroll
    for (int d = 16; d; d >>= 1) rmax = fmaxf(rmax, __shfl_xor(rmax, d, 32));

    unsigned prefix = 0u;
    for (int bit = 31; bit >= 0; --bit) {
      unsigned cand = prefix | (1u << bit);
      int cnt = 0;
      for (int j = lane; j < SEQ; j += 32) {
        unsigned u = __float_as_uint(row[j]);
        unsigned key = (u & 0x80000000u) ? ~u : (u | 0x80000000u);
        cnt += (key >= cand) ? 1 : 0;
      }
#pragma unroll
      for (int d = 16; d; d >>= 1) cnt += __shfl_xor(cnt, d, 32);
      if (cnt >= KTOP) prefix = cand;
    }
    unsigned tb = (prefix & 0x80000000u) ? (prefix & 0x7FFFFFFFu) : ~prefix;
    float thr = __uint_as_float(tb);

    float den = 0.f;
    for (int j = lane; j < SEQ; j += 32) {
      float v = row[j];
      if (v >= thr) den += __expf(v - rmax);
    }
#pragma unroll
    for (int d = 16; d; d >>= 1) den += __shfl_xor(den, d, 32);
    float inv = 1.0f / den;

    for (int j = lane; j < SEQ; j += 32) {
      float v = row[j];
      row[j] = (v >= thr) ? __expf(v - rmax) * inv : 0.0f;
    }
  }
  __syncthreads();

  // Phase 2: out = P @ V, dense WMMA over masked P. Wave w owns d-subtile w.
  // Vt chunks are staged into LDS with CDNA5 async global->LDS copies.
  v8f o = {};
  const half_t* vsrc = Vt + (size_t)bh * DHEAD * SEQ;
  for (int kc = 0; kc < SEQ / 32; ++kc) {
    const int kbase = kc * 32;
    // 256 x 16B segments: [64 d][4 segs of 8 halves]; 2 per thread.
#pragma unroll
    for (int it = 0; it < 2; ++it) {
      int seg = threadIdx.x + it * 128;
      int d   = seg >> 2;
      int off = (seg & 3) * 8;
      async_copy16(vt + d * 32 + off, vsrc + (size_t)d * SEQ + kbase + off);
    }
    wait_async();
    __syncthreads();

    const float* prow = sc + (size_t)ln * SEQ + kbase;
    v16h pa = make_a_f32(prow, half_id * 8);          // probs -> f16 A operand
    const half_t* vrow = vt + (16 * wave + ln) * 32;  // B: lane=d col, K=k
    v16h vb = make_b_f16(vrow, half_id * 8);
    o = wmma_f16(pa, vb, o);
    __syncthreads();
  }

  const int dcol = 16 * wave + ln;
#pragma unroll
  for (int r = 0; r < 8; ++r) {
    int mm = b * SEQ + qbase + r + 8 * half_id;
    AO[(size_t)mm * HDIM + head * DHEAD + dcol] = (half_t)o[r];
  }
}

// ---------------------------------------------------------------------------
// Kernel 3: output projection  out = AO @ Wo^T + bo   (f32 result)
// grid = (H/64, M/64); block = 128.
// ---------------------------------------------------------------------------
__global__ void out_proj_kernel(const half_t* __restrict__ AO,
                                const half_t* __restrict__ Woh,
                                const float* __restrict__ bo,
                                float* __restrict__ out) {
  const int lane    = threadIdx.x & 31;
  const int wave    = threadIdx.x >> 5;
  const int half_id = lane >> 4;
  const int ln      = lane & 15;

  const int mbase = blockIdx.y * 64 + wave * 16;
  const int nbase = blockIdx.x * 64;

  const half_t* arow = AO + (size_t)(mbase + ln) * HDIM;
  v8f acc[4] = {};

  for (int hb = 0; hb < HDIM; hb += 32) {
    __builtin_prefetch(arow + hb + 64, 0, 0);
    v16h a = make_a_f16(arow, hb + half_id * 8);
#pragma unroll
    for (int t = 0; t < 4; ++t) {
      const half_t* wrow = Woh + (size_t)(nbase + 16 * t + ln) * HDIM;
      v16h b = make_b_f16(wrow, hb + half_id * 8);
      acc[t] = wmma_f16(a, b, acc[t]);
    }
  }

#pragma unroll
  for (int t = 0; t < 4; ++t) {
    int n = nbase + 16 * t + ln;
    float bv = bo[n];
#pragma unroll
    for (int r = 0; r < 8; ++r) {
      int mm = mbase + r + 8 * half_id;
      out[(size_t)mm * HDIM + n] = acc[t][r] + bv;
    }
  }
}

// ---------------------------------------------------------------------------
extern "C" void kernel_launch(void* const* d_in, const int* in_sizes, int n_in,
                              void* d_out, int out_size, void* d_ws,
                              size_t ws_size, hipStream_t stream) {
  (void)in_sizes; (void)n_in; (void)out_size; (void)ws_size;
  const float* x  = (const float*)d_in[0];
  const float* Wq = (const float*)d_in[1];
  const float* Wk = (const float*)d_in[2];
  const float* Wv = (const float*)d_in[3];
  const float* Wo = (const float*)d_in[4];
  const float* bo = (const float*)d_in[5];
  float* out = (float*)d_out;

  const size_t M  = 2 * SEQ;                     // 4096 rows
  const size_t XE = M * HDIM;                    // 4M elements
  const size_t WE = (size_t)HDIM * HDIM;         // 1M elements

  half_t* xh  = (half_t*)d_ws;                   // [M][H]
  half_t* Wqh = xh  + XE;
  half_t* Wkh = Wqh + WE;
  half_t* Wvh = Wkh + WE;
  half_t* Woh = Wvh + WE;
  half_t* Qh  = Woh + WE;                        // [M][H]
  half_t* Kh  = Qh  + XE;                        // [M][H]
  half_t* Vt  = Kh  + XE;                        // [B*NH][D][S]
  half_t* AO  = Vt  + XE;                        // [M][H]   (~48 MB total)

  dim3 blk(128);

  // 0) one-shot f32 -> f16 conversions (removes all cvt from GEMM loops)
  {
    dim3 cb(256);
    hipLaunchKernelGGL(cvt_f16_kernel, dim3((unsigned)(XE / 8 / 256)), cb, 0,
                       stream, x, xh, (int)(XE / 8));
    hipLaunchKernelGGL(cvt_f16_kernel, dim3((unsigned)(WE / 8 / 256)), cb, 0,
                       stream, Wq, Wqh, (int)(WE / 8));
    hipLaunchKernelGGL(cvt_f16_kernel, dim3((unsigned)(WE / 8 / 256)), cb, 0,
                       stream, Wk, Wkh, (int)(WE / 8));
    hipLaunchKernelGGL(cvt_f16_kernel, dim3((unsigned)(WE / 8 / 256)), cb, 0,
                       stream, Wv, Wvh, (int)(WE / 8));
    hipLaunchKernelGGL(cvt_f16_kernel, dim3((unsigned)(WE / 8 / 256)), cb, 0,
                       stream, Wo, Woh, (int)(WE / 8));
  }

  // 1) Q/K/V projections (z selects weight + store layout)
  dim3 g1(HDIM / 64, M / 64, 3);
  hipLaunchKernelGGL(proj_qkv_kernel, g1, blk, 0, stream,
                     xh, Wqh, Wkh, Wvh, Qh, Kh, Vt);

  // 2) fused sparse attention
  dim3 g2(SEQ / 16, 2 * NHEAD);
  size_t lds = (size_t)16 * SEQ * sizeof(float) + 64 * 32 * sizeof(half_t);
  hipLaunchKernelGGL(sparse_attn_kernel, g2, blk, lds, stream,
                     Qh, Kh, Vt, AO);

  // 3) output projection + bias
  dim3 g3(HDIM / 64, M / 64);
  hipLaunchKernelGGL(out_proj_kernel, g3, blk, 0, stream, AO, Woh, bo, out);
}
